// CTCLayer_63419487093329
// MI455X (gfx1250) — matrix operand order repaired and checked by reading, not compiled, not run on hardware
//
#include <hip/hip_runtime.h>
#include <cstdint>

// ---------------------------------------------------------------------------
// CTC batch cost for MI455X (gfx1250, wave32).
// One workgroup per batch row b. 288 threads = 9 wave32.
//   Phase A: TDM streams y_pred[b, t0:t0+16, 0:128] (8KB) into LDS (dbl buf).
//   Phase B: threads compute lp = log(p+eps), split into f16 hi/lo planes.
//   Phase C: WMMA one-hot gather lp[s,t'] = onehot[s,:] x logp[:,t'] (f32 acc,
//            hi+lo f16 split => near-exact), D scattered to LDS.
//   Phase D: alpha scan, thread s owns state s, LDS double-buffered alpha,
//            one barrier per timestep.
// ---------------------------------------------------------------------------

typedef __attribute__((ext_vector_type(16))) _Float16     v16h;
typedef __attribute__((ext_vector_type(8)))  float        v8f;
typedef __attribute__((ext_vector_type(4)))  unsigned int u32x4;
typedef __attribute__((ext_vector_type(8)))  int          i32x8;
typedef __attribute__((ext_vector_type(4)))  int          i32x4;

#define CTC_NEG  (-1e9f)
#define CTC_EPS  (1e-7f)

#if defined(__has_builtin)
#  if __has_builtin(__builtin_amdgcn_tensor_load_to_lds) && __has_builtin(__builtin_amdgcn_s_wait_tensorcnt)
#    define CTC_HAVE_TDM 1
#  else
#    define CTC_HAVE_TDM 0
#  endif
#else
#  define CTC_HAVE_TDM 0
#endif

__device__ __forceinline__ float lae2(float a, float b) {
  float m = fmaxf(a, b);
  return m + __logf(__expf(a - m) + __expf(b - m));
}
__device__ __forceinline__ float lae3(float a, float b, float c) {
  float m = fmaxf(fmaxf(a, b), c);
  return m + __logf(__expf(a - m) + __expf(b - m) + __expf(c - m));
}

#if CTC_HAVE_TDM
// Issue a TDM 2D tile load: 16 rows x 128 cols of f32 (contiguous) -> LDS.
// D# per CDNA5 ISA.
__device__ __forceinline__ void tdm_load_tile(unsigned int lds_byte_addr, const float* gsrc) {
  unsigned long long ga = (unsigned long long)(uintptr_t)gsrc;
  u32x4 g0;
  g0[0] = 1u;                                           // count=1 (valid), user mode
  g0[1] = lds_byte_addr;                                // lds_addr
  g0[2] = (unsigned int)(ga & 0xffffffffu);             // global_addr[31:0]
  g0[3] = (unsigned int)(ga >> 32) | 0x80000000u;       // global_addr[56:32] | type=2
  i32x8 g1;
  g1[0] = 0x00020000;                                   // data_size=2 (4B), mask=0
  g1[1] = (int)(128u << 16);                            // tensor_dim0 = 128 (bits 79:48)
  g1[2] = (int)(16u << 16);                             // tensor_dim1 = 16  (bits 111:80)
  g1[3] = (int)(128u << 16);                            // tile_dim0   = 128 (bits 127:112)
  g1[4] = 16;                                           // tile_dim1   = 16  (bits 143:128)
  g1[5] = 128;                                          // tensor_dim0_stride = 128
  g1[6] = 0;
  g1[7] = 0;
  i32x4 z4 = {0, 0, 0, 0};
#if __has_include(<hip/amd_detail/amd_gfx1250_TDM.h>)
  i32x8 z8 = {0, 0, 0, 0, 0, 0, 0, 0};
  __builtin_amdgcn_tensor_load_to_lds(g0, g1, z4, z4, z8, 0);   // clang-23 6-arg form
#else
  __builtin_amdgcn_tensor_load_to_lds(g0, g1, z4, z4, 0);       // ROCm 7.2 5-arg form
#endif
}
#endif

__global__ void __launch_bounds__(288, 1)
ctc_batch_cost_kernel(const int* __restrict__ y_true,
                      const float* __restrict__ y_pred,
                      float* __restrict__ out) {
  constexpr int T  = 1024, C = 128, L = 128;
  constexpr int S  = 2 * L + 1;          // 257
  constexpr int TS = 16;                 // timesteps per WMMA tile (N dim)
  constexpr int NSTEP = T / TS;          // 64
  constexpr int SP  = 288;               // padded states (= blockDim)
  constexpr int SPL = 292;               // lp row stride (bank-conflict pad)
  constexpr int BLANK = C - 1;

  __shared__ float    raw[2][TS * C];    // TDM destination, double buffered (16 KB)
  __shared__ _Float16 bst[2][C * TS];    // hi / lo f16 planes of logp, [N][K] (8 KB)
  __shared__ float    lp_lds[TS * SPL];  // gathered lp [t'][s] (18.7 KB)
  __shared__ float    abuf[2][SP + 2];   // alpha dbl buffer, +2 front pad (2.3 KB)
  __shared__ int      ext_s[SP];
  __shared__ int      skip_s[SP];

  const int tid  = threadIdx.x;
  const int lane = tid & 31;
  const int wid  = tid >> 5;             // 0..8
  const int b    = blockIdx.x;

  // --- extended label sequence: blank at even s, labels at odd s -------------
  {
    int e;
    if (tid < S) e = (tid & 1) ? y_true[b * L + ((tid - 1) >> 1)] : BLANK;
    else         e = 0xFFFF;             // sentinel: matches no channel
    ext_s[tid] = e;
  }
  if (tid < 2) { abuf[0][tid] = CTC_NEG; abuf[1][tid] = CTC_NEG; }
  __syncthreads();
  // skip (s-2 transition allowed): own-index write/read only, no extra barrier
  skip_s[tid] = (tid < S) && (tid & 1) && (tid < 2 || ext_s[tid] != ext_s[tid - 2]);

  // --- build one-hot A fragments (16-bit A 16x32: m = lane&15, contiguous k = 16*half + j) --------
  // wave w owns M-tiles 2w, 2w+1 (17 real tiles over 9 waves; tile 17 is pad).
  v16h afr[2][4];
#pragma unroll
  for (int ti = 0; ti < 2; ++ti) {
    const int tile = 2 * wid + ti;                 // 0..17
    const int m    = tile * 16 + (lane & 15);      // state row, < 288
    const int half = lane >> 4;
    const int e    = ext_s[m];
#pragma unroll
    for (int kc = 0; kc < 4; ++kc) {
      v16h a;
#pragma unroll
      for (int j = 0; j < 16; ++j) {
        const int kl = j + half * 16;   // lane/slot -> K
        a[j] = (e == kc * 32 + kl) ? (_Float16)1.0f : (_Float16)0.0f;
      }
      afr[ti][kc] = a;
    }
  }

  const int  s      = tid;
  const bool active = (s < S);
  const int  skipv  = skip_s[tid];

  const float* gbase = y_pred + (size_t)b * T * C;

#if CTC_HAVE_TDM
  if (wid == 0) tdm_load_tile((unsigned int)(uintptr_t)(&raw[0][0]), gbase);
#endif

  float alphaReg = CTC_NEG;
  int   cur      = 0;

  for (int step = 0; step < NSTEP; ++step) {
    const int t0 = step * TS;
    const int rb = step & 1;

#if CTC_HAVE_TDM
    if (wid == 0) __builtin_amdgcn_s_wait_tensorcnt(0);   // this step's tile landed
    __syncthreads();
    if (wid == 0 && step + 1 < NSTEP)                     // overlap next DMA w/ compute
      tdm_load_tile((unsigned int)(uintptr_t)(&raw[rb ^ 1][0]),
                    gbase + (size_t)(t0 + TS) * C);
#else
    for (int i = tid; i < TS * C; i += SP)
      raw[rb][i] = gbase[(size_t)t0 * C + i];
    __syncthreads();
#endif

    // --- stage: log(p+eps), f16 hi/lo split, into [N=t'][K=c] ------
    for (int i = tid; i < TS * C; i += SP) {
      const int c  = i >> 4;
      const int tt = i & 15;
      const float lpv = __logf(raw[rb][tt * C + c] + CTC_EPS);
      const _Float16 hi = (_Float16)lpv;
      const _Float16 lo = (_Float16)(lpv - (float)hi);
      bst[0][tt * C + c] = hi;
      bst[1][tt * C + c] = lo;
    }
    __syncthreads();

    // --- WMMA one-hot gather: D[s,t'] = onehot(SxC) x logp(Cx16) -------------
#pragma unroll
    for (int ti = 0; ti < 2; ++ti) {
      const int tile = 2 * wid + ti;
      if (tile < 17) {                                    // wave-uniform branch
        v8f acc = {};
#pragma unroll
        for (int kc = 0; kc < 4; ++kc) {
          const int krow = kc * 32 + (lane >> 4) * 16;    // B layout: lane&15 <-> N, slot <-> K
          v16h bh = *(const v16h*)(&bst[0][(lane & 15) * C + krow]);
          acc = __builtin_amdgcn_wmma_f32_16x16x32_f16(false, afr[ti][kc], false, bh,
                                                       (short)0, acc, false, false);
          v16h bl = *(const v16h*)(&bst[1][(lane & 15) * C + krow]);
          acc = __builtin_amdgcn_wmma_f32_16x16x32_f16(false, afr[ti][kc], false, bl,
                                                       (short)0, acc, false, false);
        }
        // D layout: VGPR j -> rows j (lanes 0-15) / j+8 (lanes 16-31), N = lane&15
        const int col   = lane & 15;
        const int rbase = (lane < 16) ? 0 : 8;
        float* dst = &lp_lds[col * SPL + tile * 16 + rbase];
        *(float4*)(dst)     = make_float4(acc[0], acc[1], acc[2], acc[3]);
        *(float4*)(dst + 4) = make_float4(acc[4], acc[5], acc[6], acc[7]);
      }
    }
    __syncthreads();

    // --- alpha scan: 16 timesteps, one barrier each --------------------------
    for (int tt = 0; tt < TS; ++tt) {
      const int gt = t0 + tt;
      if (active) {
        const float lpv = lp_lds[tt * SPL + s];
        float na;
        if (gt == 0) {
          na = (s < 2) ? lpv : CTC_NEG;
        } else {
          const float a1 = abuf[cur][2 + s - 1];
          const float a2 = skipv ? abuf[cur][2 + s - 2] : CTC_NEG;
          na = lae3(alphaReg, a1, a2) + lpv;
        }
        abuf[cur ^ 1][2 + s] = na;
        alphaReg = na;
      }
      __syncthreads();
      cur ^= 1;
    }
  }

  if (tid == 0)
    out[b] = -lae2(abuf[cur][2 + S - 1], abuf[cur][2 + S - 2]);
}

extern "C" void kernel_launch(void* const* d_in, const int* in_sizes, int n_in,
                              void* d_out, int out_size, void* d_ws, size_t ws_size,
                              hipStream_t stream) {
  (void)in_sizes; (void)n_in; (void)out_size; (void)d_ws; (void)ws_size;
  const int*   y_true = (const int*)d_in[0];
  const float* y_pred = (const float*)d_in[1];
  float*       out    = (float*)d_out;
  hipLaunchKernelGGL(ctc_batch_cost_kernel, dim3(256), dim3(288), 0, stream,
                     y_true, y_pred, out);
}
